// LSTMModel_84490596647594
// MI455X (gfx1250) — compile-verified, matrix operand order checked by reading, not run on hardware
//
#include <hip/hip_runtime.h>

typedef __attribute__((ext_vector_type(16))) _Float16 v16h;
typedef __attribute__((ext_vector_type(8)))  float    v8f;
typedef __attribute__((ext_vector_type(4)))  float    v4f;
typedef __attribute__((ext_vector_type(4)))  _Float16 v4h;

#define HID      64
#define FORECAST 30
#define ROWS     16           // batch rows per workgroup
#define GSTRIDE  260          // padded gates row stride (floats)  -> bank de-conflict
#define HSTRIDE  72           // padded h row stride (halves), 144B = 16B-aligned rows

union AFrag { v16h v; v4f f4[2]; };

// ---- branchless fast transcendentals on the CDNA5 TRANS pipe ----
__device__ __forceinline__ float fast_tanh(float x) {
#if __has_builtin(__builtin_amdgcn_tanh_f32)
    return __builtin_amdgcn_tanh_f32(x);          // native v_tanh_f32
#elif __has_builtin(__builtin_amdgcn_tanhf)
    return __builtin_amdgcn_tanhf(x);
#else
    // tanh(x) = 1 - 2/(exp2(2*log2e*x)+1) : 1 trans(exp) + 1 trans(rcp), branchless
    const float e = __builtin_amdgcn_exp2f(x * 2.8853900817779268f);
    return __builtin_fmaf(-2.0f, __builtin_amdgcn_rcpf(e + 1.0f), 1.0f);
#endif
}

__device__ __forceinline__ float fast_sigmoid(float x) {
    // sigma(x) = 0.5 + 0.5*tanh(0.5x) : branchless, 1 trans op
    return __builtin_fmaf(0.5f, fast_tanh(0.5f * x), 0.5f);
}

__global__ __launch_bounds__(256, 2)
void lstm_forecast_kernel(const float* __restrict__ x,
                          const float* __restrict__ W_ih,
                          const float* __restrict__ W_hh,
                          const float* __restrict__ b_ih,
                          const float* __restrict__ b_hh,
                          const float* __restrict__ fc_W,
                          const float* __restrict__ fc_b,
                          float* __restrict__ out,
                          int T) {
    __shared__ float    gates[ROWS * GSTRIDE];   // ~16.6 KB
    __shared__ _Float16 hbuf[ROWS * HSTRIDE];    // ~2.3 KB

    const int tid  = threadIdx.x;
    const int wv   = tid >> 5;          // wave 0..7 : owns gate cols [wv*32, wv*32+32)
    const int lane = tid & 31;
    const int ln16 = lane & 15;
    const int hi   = lane >> 4;         // half-wave select (WMMA layout)

    // ---------------- register-resident B fragments of W_hh^T ----------------
    // B tile (K=32 x N=16), f16: lane l<16 holds N=l, K = {0..7, 16..23};
    // lanes 16..31 hold N=l-16, K = {8..15, 24..31}  (mirrors documented A layout)
    v16h bfrag[2][2];   // [n-tile][k-slice]
    #pragma unroll
    for (int nt = 0; nt < 2; ++nt) {
        const int col = wv * 32 + nt * 16 + ln16;     // gate column 0..255
        #pragma unroll
        for (int kb = 0; kb < 2; ++kb) {
            const int base0 = kb * 32 + (hi ? 8 : 0);
            const int base1 = kb * 32 + (hi ? 24 : 16);
            v16h b;
            #pragma unroll
            for (int e = 0; e < 8; ++e)
                b[e] = (_Float16)W_hh[col * HID + base0 + e];
            #pragma unroll
            for (int e = 0; e < 8; ++e)
                b[8 + e] = (_Float16)W_hh[col * HID + base1 + e];
            bfrag[nt][kb] = b;
        }
    }

    // ------------- per-thread elementwise ownership: 4 cells of one row -------------
    const int row = tid >> 4;            // 0..15  (fixed for whole sequence)
    const int hid = (tid & 15) * 4;      // 0,4,...,60
    float wih[16], bsum[16];
    #pragma unroll
    for (int g = 0; g < 4; ++g) {
        #pragma unroll
        for (int e = 0; e < 4; ++e) {
            const int col = g * HID + hid + e;
            wih[g * 4 + e]  = W_ih[col];
            bsum[g * 4 + e] = b_ih[col] + b_hh[col];
        }
    }

    const int grow0 = blockIdx.x * ROWS;
    const float* xrow = x + (size_t)(grow0 + row) * (size_t)T;

    // h = 0, c = 0
    float c_st[4] = {0.f, 0.f, 0.f, 0.f};
    v4h hz; hz[0] = (_Float16)0.f; hz[1] = (_Float16)0.f;
            hz[2] = (_Float16)0.f; hz[3] = (_Float16)0.f;
    *(v4h*)&hbuf[row * HSTRIDE + hid] = hz;
    __syncthreads();

    // A-fragment LDS offsets (halves), fixed per lane
    const int m      = ln16;
    const int aoff00 = m * HSTRIDE +  0 + (hi ? 8 : 0);
    const int aoff01 = m * HSTRIDE +  0 + (hi ? 24 : 16);
    const int aoff10 = m * HSTRIDE + 32 + (hi ? 8 : 0);
    const int aoff11 = m * HSTRIDE + 32 + (hi ? 24 : 16);

    // ------------------------------ recurrence ------------------------------
    v4f xq = *(const v4f*)xrow;          // 4 timesteps of x, prefetched
    const int nT4 = T >> 2;
    for (int t4 = 0; t4 < nT4; ++t4) {
        const v4f xcur = xq;
        const int tnext = (t4 + 1 < nT4) ? (t4 + 1) : t4;   // branchless prefetch
        xq = *(const v4f*)(xrow + (size_t)tnext * 4);

        #pragma unroll
        for (int ti = 0; ti < 4; ++ti) {
            const float xv = xcur[ti];

            // A fragments: h (f16) from LDS
            AFrag a0, a1;
            a0.f4[0] = *(const v4f*)&hbuf[aoff00];
            a0.f4[1] = *(const v4f*)&hbuf[aoff01];
            a1.f4[0] = *(const v4f*)&hbuf[aoff10];
            a1.f4[1] = *(const v4f*)&hbuf[aoff11];

            // gates_tile = h @ W_hh^T  (two chained K=32 WMMAs per N-tile)
            #pragma unroll
            for (int nt = 0; nt < 2; ++nt) {
                v8f acc = {};
                acc = __builtin_amdgcn_wmma_f32_16x16x32_f16(
                        false, a0.v, false, bfrag[nt][0], (short)0, acc, false, false);
                acc = __builtin_amdgcn_wmma_f32_16x16x32_f16(
                        false, a1.v, false, bfrag[nt][1], (short)0, acc, false, false);
                const int col = wv * 32 + nt * 16 + ln16;
                #pragma unroll
                for (int r = 0; r < 8; ++r)
                    gates[(r + hi * 8) * GSTRIDE + col] = acc[r];
            }
            __syncthreads();

            // fused elementwise LSTM cell update (adds x*W_ih + b here), branchless
            const float* grow = &gates[row * GSTRIDE];
            v4f iv = *(const v4f*)&grow[0 * HID + hid];
            v4f fv = *(const v4f*)&grow[1 * HID + hid];
            v4f gv = *(const v4f*)&grow[2 * HID + hid];
            v4f ov = *(const v4f*)&grow[3 * HID + hid];
            v4h hv;
            #pragma unroll
            for (int e = 0; e < 4; ++e) {
                const float ig = fast_sigmoid(__builtin_fmaf(xv, wih[0*4+e], iv[e] + bsum[0*4+e]));
                const float fg = fast_sigmoid(__builtin_fmaf(xv, wih[1*4+e], fv[e] + bsum[1*4+e]));
                const float gg = fast_tanh   (__builtin_fmaf(xv, wih[2*4+e], gv[e] + bsum[2*4+e]));
                const float og = fast_sigmoid(__builtin_fmaf(xv, wih[3*4+e], ov[e] + bsum[3*4+e]));
                c_st[e] = __builtin_fmaf(fg, c_st[e], ig * gg);
                hv[e]   = (_Float16)(og * fast_tanh(c_st[e]));
            }
            *(v4h*)&hbuf[row * HSTRIDE + hid] = hv;
            __syncthreads();
        }
    }

    // ------------------------------ final FC ------------------------------
    // out[16,30] = h_last @ fc_W^T + fc_b
    for (int idx = tid; idx < ROWS * FORECAST; idx += 256) {
        const int r = idx / FORECAST;
        const int j = idx - r * FORECAST;
        float s = fc_b[j];
        #pragma unroll 8
        for (int k = 0; k < HID; ++k)
            s += (float)hbuf[r * HSTRIDE + k] * fc_W[j * HID + k];
        out[(size_t)(grow0 + r) * FORECAST + j] = s;
    }
}

extern "C" void kernel_launch(void* const* d_in, const int* in_sizes, int n_in,
                              void* d_out, int out_size, void* d_ws, size_t ws_size,
                              hipStream_t stream) {
    const float* x    = (const float*)d_in[0];
    const float* W_ih = (const float*)d_in[1];
    const float* W_hh = (const float*)d_in[2];
    const float* b_ih = (const float*)d_in[3];
    const float* b_hh = (const float*)d_in[4];
    const float* fc_W = (const float*)d_in[5];
    const float* fc_b = (const float*)d_in[6];
    float* out = (float*)d_out;

    const int T = 1024;
    const int B = in_sizes[0] / T;       // x is [B, T, 1]

    dim3 grid(B / ROWS), block(256);
    hipLaunchKernelGGL(lstm_forecast_kernel, grid, block, 0, stream,
                       x, W_ih, W_hh, b_ih, b_hh, fc_W, fc_b, out, T);
}
